// Encoder_3_48593259987150
// MI455X (gfx1250) — matrix operand order repaired and checked by reading, not compiled, not run on hardware
//
#include <hip/hip_runtime.h>
#include <hip/hip_bf16.h>

// ---------------------------------------------------------------------------
// LSTM final-state kernel for MI455X (gfx1250), wave32 + bf16 WMMA.
// B=64, S=1024, D=256, H=800, 4H=3200.
// XG is stored in wave-fragment-major tile layout so producer (xproj) and
// consumer (scan) use the identical WMMA C-fragment convention.
// ---------------------------------------------------------------------------

typedef __attribute__((ext_vector_type(16))) __bf16 v16bf;
typedef __attribute__((ext_vector_type(8)))  __bf16 v8bf;
typedef __attribute__((ext_vector_type(8)))  float  v8f;

#define LSTM_B 64
#define LSTM_S 1024
#define LSTM_D 256
#define LSTM_H 800
#define GATES4 3200   // 4*H
#define NTILE  800    // 4 btiles * 4 gates * 50 htiles

// workspace layout (bytes)
#define WS_CTR  0u
#define WS_HBF  256u                               // 2 * 64 * 800 bf16 = 204800
#define WS_WXC  205056u                            // 3200*256 bf16 = 1638400
#define WS_WHC  1843456u                           // 3200*800 bf16 = 5120000
#define WS_XG   6963456u                           // 1024*800*256 f32 = 838860800

__device__ __forceinline__ __bf16 f2bf(float f) {
  unsigned u = __builtin_bit_cast(unsigned, f);
  unsigned r = u + 0x7fffu + ((u >> 16) & 1u);      // round-to-nearest-even
  unsigned short h = (unsigned short)(r >> 16);
  return __builtin_bit_cast(__bf16, h);
}

__device__ __forceinline__ float fast_sigmoid(float x) {
  return 1.0f / (1.0f + __expf(-x));
}
__device__ __forceinline__ float fast_tanh(float x) {
  return 1.0f - 2.0f / (__expf(2.0f * x) + 1.0f);
}

__device__ __forceinline__ v8f wmma_bf16(v16bf a, v16bf b, v8f c) {
  // D(f32 16x16) = A(16x32 bf16) * B(32x16 bf16) + C
  return __builtin_amdgcn_wmma_f32_16x16x32_bf16(
      /*neg_a=*/false, a, /*neg_b=*/false, b,
      /*c_mod=*/(short)0, c, /*reuse_a=*/false, /*reuse_b=*/false);
}

// CDNA5 async memory->LDS copy (ASYNCcnt-tracked), per CDNA5 ISA.
__device__ __forceinline__ void async_load_b128(unsigned lds_off, const void* gaddr) {
  asm volatile("global_load_async_to_lds_b128 %0, %1, off"
               :
               : "v"(lds_off), "v"(gaddr)
               : "memory");
}
__device__ __forceinline__ void wait_asynccnt0() {
  asm volatile("s_wait_asynccnt 0" ::: "memory");
}

// ---------------------------------------------------------------------------
// Phase 0: transpose + convert weights to bf16 column-major [col][k] so that
// B-fragments in this code's k order (per-lane: one column, 16 contiguous K values) are single
// contiguous 32-byte loads.
// ---------------------------------------------------------------------------
__global__ void cvt_wx(const float* __restrict__ Wx, __bf16* __restrict__ Wxc) {
  int idx = blockIdx.x * blockDim.x + threadIdx.x;   // 3200*256 exact
  int c = idx / LSTM_D, k = idx % LSTM_D;
  Wxc[idx] = f2bf(Wx[(size_t)k * GATES4 + c]);
}

__global__ void cvt_wh(const float* __restrict__ Wh, __bf16* __restrict__ Whc) {
  int idx = blockIdx.x * blockDim.x + threadIdx.x;   // 3200*800 exact
  int c = idx / LSTM_H, k = idx % LSTM_H;
  Whc[idx] = f2bf(Wh[(size_t)k * GATES4 + c]);
}

// ---------------------------------------------------------------------------
// Phase 1: per timestep s, one GEMM (64 x 256) x (256 x 3200):
//   XG(s)[b, g*H+h] = bias + (x[b,s,:]*mask[g,b,:]) . Wx[:, g*H+h]
// Grid: (1024 s, 4 btile). Block 128 (wave w = gate w). Output stored in
// fragment-major tiles: XG[s][tile][lane][v], tile = (btile*4+g)*50+htile.
// ---------------------------------------------------------------------------
__global__ __launch_bounds__(128)
void lstm_xproj(const float* __restrict__ x, const float* __restrict__ masks,
                const float* __restrict__ bias, const __bf16* __restrict__ Wxc,
                float* __restrict__ XG) {
  extern __shared__ char smem1[];
  float* lds_x = (float*)smem1;                 // [16][256]  (b rows)
  float* lds_m = lds_x + 16 * LSTM_D;           // [4][16][256]

  const int s   = blockIdx.x;
  const int bt  = blockIdx.y;                   // b-tile
  const int b0  = bt * 16;
  const int tid = threadIdx.x;

  // x rows: x[b0+r][s][0..256) ; 16 rows of 64 float4
  for (int i = tid; i < 16 * (LSTM_D / 4); i += 128) {
    const int r = i >> 6, c4 = i & 63;
    ((float4*)lds_x)[i] =
        *(const float4*)(x + ((size_t)(b0 + r) * LSTM_S + s) * LSTM_D + c4 * 4);
  }
  // masks[g][b0+r][0..256) ; 64 rows of 64 float4
  for (int i = tid; i < 64 * (LSTM_D / 4); i += 128) {
    const int row = i >> 6, c4 = i & 63;
    const int g = row >> 4, r = row & 15;
    ((float4*)lds_m)[i] =
        *(const float4*)(masks + ((size_t)g * LSTM_B + b0 + r) * LSTM_D + c4 * 4);
  }
  __syncthreads();

  const int wave = tid >> 5;        // gate index
  const int lane = tid & 31;
  const int half = lane >> 4;       // K-half selector: k = 16*half + s in each 32-chunk
  const int n    = lane & 15;
  const float* xrow = lds_x + (size_t)n * LSTM_D;               // row M = b offset
  const float* mrow = lds_m + ((size_t)wave * 16 + n) * LSTM_D;

  // A fragments (16x32 bf16 per K-chunk), built with vector LDS reads.
  v16bf afrag[8];
#pragma unroll
  for (int kc = 0; kc < 8; ++kc) {
    const int base = kc * 32 + half * 16;
    const v8f xl = *(const v8f*)(xrow + base);
    const v8f xh = *(const v8f*)(xrow + base + 8);
    const v8f ml = *(const v8f*)(mrow + base);
    const v8f mh = *(const v8f*)(mrow + base + 8);
    v16bf a;
#pragma unroll
    for (int j = 0; j < 8; ++j) {
      a[j]     = f2bf(xl[j] * ml[j]);
      a[8 + j] = f2bf(xh[j] * mh[j]);
    }
    afrag[kc] = a;
  }

  for (int ht = 0; ht < LSTM_H / 16; ++ht) {
    const int col0 = wave * LSTM_H + ht * 16;
    v8f acc = {};
#pragma unroll
    for (int kc = 0; kc < 8; ++kc) {
      const v16bf bf =
          *(const v16bf*)(Wxc + (size_t)(col0 + n) * LSTM_D + kc * 32 + half * 16);
      acc = wmma_bf16(afrag[kc], bf, acc);
    }
    const float bv = bias[col0 + n];
#pragma unroll
    for (int v = 0; v < 8; ++v) acc[v] += bv;
    const int tile = (bt * 4 + wave) * 50 + ht;
    *(v8f*)(XG + ((size_t)s * NTILE + tile) * 256 + (size_t)lane * 8) = acc;
  }
}

// ---------------------------------------------------------------------------
// Phase 2: persistent sequential scan. 50 WGs (one 16-wide h-tile each) x
// 4 waves (one 16-row b-tile each). Wh slice (102 KB bf16) resident in LDS
// for all 1024 steps; h (100 KB bf16) refilled per step with async-to-LDS;
// c tile lives in VGPRs. Grid barrier = monotonic atomic counter.
// ---------------------------------------------------------------------------
__global__ __launch_bounds__(128)
void lstm_scan(const float* __restrict__ XG, const __bf16* __restrict__ Whc,
               __bf16* __restrict__ hbf, unsigned* __restrict__ ctr,
               float* __restrict__ out) {
  extern __shared__ char smem[];
  __bf16* lds_h  = (__bf16*)smem;                 // [64][800]  at LDS offset 0
  __bf16* lds_wh = lds_h + LSTM_B * LSTM_H;       // [4][16][800]

  const int htile = blockIdx.x;
  const int tid   = threadIdx.x;
  const int wave  = tid >> 5;     // b-tile
  const int lane  = tid & 31;
  const int half  = lane >> 4;
  const int n     = lane & 15;
  const unsigned NWG = gridDim.x;

  // one-time Wh slice load: cols [g*800 + htile*16, +16) are contiguous in Whc
  for (int g = 0; g < 4; ++g) {
    const unsigned* src =
        (const unsigned*)(Whc + (size_t)(g * LSTM_H + htile * 16) * LSTM_H);
    unsigned* dst = (unsigned*)(lds_wh + (size_t)g * 16 * LSTM_H);
    for (int i = tid; i < 16 * LSTM_H / 2; i += 128) dst[i] = src[i];
  }
  __syncthreads();

  v8f c_acc = {};
  const __bf16* arow = lds_h + (size_t)(wave * 16 + n) * LSTM_H;
  const __bf16* brow = lds_wh + (size_t)n * LSTM_H;

  for (int t = 0; t < LSTM_S; ++t) {
    // refill h_t into LDS via CDNA5 async memory->LDS (ASYNCcnt-tracked)
    {
      const char* src = (const char*)(hbf + (size_t)(t & 1) * LSTM_B * LSTM_H);
      for (int i = tid; i < LSTM_B * LSTM_H * 2 / 16; i += 128)
        async_load_b128((unsigned)(i * 16), src + (size_t)i * 16);
      wait_asynccnt0();
    }
    __syncthreads();

    // C-init from fragment-major XG: one 32B vector load per gate
    const float* xg_t = XG + (size_t)t * NTILE * 256;
    v8f acc[4];
#pragma unroll
    for (int g = 0; g < 4; ++g) {
      const int tile = (wave * 4 + g) * 50 + htile;
      acc[g] = *(const v8f*)(xg_t + (size_t)tile * 256 + (size_t)lane * 8);
    }
    if (t + 1 < LSTM_S) {   // pull next step's C-init tiles toward the caches
      const float* nxt = XG + (size_t)(t + 1) * NTILE * 256 + (size_t)lane * 8;
#pragma unroll
      for (int g = 0; g < 4; ++g)
        __builtin_prefetch(nxt + (size_t)((wave * 4 + g) * 50 + htile) * 256, 0, 1);
    }

    for (int kc = 0; kc < LSTM_H / 32; ++kc) {
      v16bf a;
      {
        const v8bf lo = *(const v8bf*)(arow + kc * 32 + half * 16);
        const v8bf hi = *(const v8bf*)(arow + kc * 32 + half * 16 + 8);
#pragma unroll
        for (int j = 0; j < 8; ++j) { a[j] = lo[j]; a[8 + j] = hi[j]; }
      }
#pragma unroll
      for (int g = 0; g < 4; ++g) {
        const v16bf bb =
            *(const v16bf*)(brow + (size_t)g * 16 * LSTM_H + kc * 32 + half * 16);
        acc[g] = wmma_bf16(a, bb, acc[g]);
      }
    }

    // gate nonlinearity + state update; c stays in VGPRs
    __bf16* hnext = hbf + (size_t)((t + 1) & 1) * LSTM_B * LSTM_H;
    const int brow_m = wave * 16 + half * 8;
    const int hcol   = htile * 16 + n;
#pragma unroll
    for (int v = 0; v < 8; ++v) {
      const float iv = fast_sigmoid(acc[0][v]);
      const float fv = fast_sigmoid(acc[1][v]);
      const float gv = fast_tanh(acc[2][v]);
      const float ov = fast_sigmoid(acc[3][v]);
      const float cv = fv * c_acc[v] + iv * gv;
      c_acc[v] = cv;
      const float hv = ov * fast_tanh(cv);
      hnext[(size_t)(brow_m + v) * LSTM_H + hcol] = f2bf(hv);
      if (t == LSTM_S - 1) {
        out[(size_t)(brow_m + v) * LSTM_H + hcol] = hv;                           // final h
        out[(size_t)LSTM_B * LSTM_H + (size_t)(brow_m + v) * LSTM_H + hcol] = cv; // final c
      }
    }
    __syncthreads();

    // device-wide barrier: monotonic counter, target = NWG*(t+1)
    if (tid == 0) {
      __hip_atomic_fetch_add(ctr, 1u, __ATOMIC_RELEASE, __HIP_MEMORY_SCOPE_AGENT);
      const unsigned target = NWG * (unsigned)(t + 1);
      while (__hip_atomic_load(ctr, __ATOMIC_ACQUIRE, __HIP_MEMORY_SCOPE_AGENT) < target)
        __builtin_amdgcn_s_sleep(1);
    }
    __syncthreads();
    __builtin_amdgcn_fence(__ATOMIC_ACQUIRE, "agent");
  }
}

// ---------------------------------------------------------------------------
extern "C" void kernel_launch(void* const* d_in, const int* in_sizes, int n_in,
                              void* d_out, int out_size, void* d_ws, size_t ws_size,
                              hipStream_t stream) {
  (void)in_sizes; (void)n_in; (void)out_size; (void)ws_size;
  const float* x     = (const float*)d_in[0];   // [64][1024][256]
  const float* Wx    = (const float*)d_in[1];   // [256][3200]
  const float* Wh    = (const float*)d_in[2];   // [800][3200]
  const float* bias  = (const float*)d_in[3];   // [3200]
  const float* masks = (const float*)d_in[4];   // [4][64][256]

  char* ws = (char*)d_ws;
  unsigned* ctr = (unsigned*)(ws + WS_CTR);
  __bf16*   hbf = (__bf16*)(ws + WS_HBF);
  __bf16*   Wxc = (__bf16*)(ws + WS_WXC);
  __bf16*   Whc = (__bf16*)(ws + WS_WHC);
  float*    XG  = (float*)(ws + WS_XG);

  // zero barrier counter + h double-buffer (h0 = 0)
  hipMemsetAsync(ws, 0, WS_WXC, stream);

  cvt_wx<<<GATES4 * LSTM_D / 256, 256, 0, stream>>>(Wx, Wxc);
  cvt_wh<<<GATES4 * LSTM_H / 256, 256, 0, stream>>>(Wh, Whc);

  const size_t xproj_lds = (size_t)(16 * LSTM_D + 4 * 16 * LSTM_D) * sizeof(float);
  hipFuncSetAttribute((const void*)lstm_xproj,
                      hipFuncAttributeMaxDynamicSharedMemorySize, (int)xproj_lds);
  lstm_xproj<<<dim3(LSTM_S, LSTM_B / 16), 128, xproj_lds, stream>>>(x, masks, bias,
                                                                    Wxc, XG);

  const size_t scan_lds = (size_t)(LSTM_B * LSTM_H + 4 * 16 * LSTM_H) * sizeof(__bf16);
  hipFuncSetAttribute((const void*)lstm_scan,
                      hipFuncAttributeMaxDynamicSharedMemorySize, (int)scan_lds);
  lstm_scan<<<LSTM_H / 16, 128, scan_lds, stream>>>(XG, Whc, hbf, ctr, (float*)d_out);
}
